// LSTMCell_88356067213735
// MI455X (gfx1250) — matrix operand order repaired and checked by reading, not compiled, not run on hardware
//
#include <hip/hip_runtime.h>

// Problem constants (fixed by the reference).
#define BDIM 8192
#define IDIM 1024
#define HDIM 1024
#define CDIM 2048   // I + H
#define MT   128    // batch rows per block (main kernel): 4 row-groups x 32 rows
#define NT   32     // gate columns per block (per gate)
#define KT   32     // K step per WMMA
#define MTF  128    // batch rows per block (fallback kernel)

typedef __bf16 v16bf __attribute__((ext_vector_type(16)));
typedef __bf16 v8bf  __attribute__((ext_vector_type(8)));
typedef float  v8f   __attribute__((ext_vector_type(8)));
typedef float  v4f   __attribute__((ext_vector_type(4)));
typedef float  v2f   __attribute__((ext_vector_type(2)));

// fp32 -> bf16 round-to-nearest-even (bit trick; no dependence on cvt ISA support)
static __device__ __forceinline__ __bf16 f2bf(float f) {
  union { float f; unsigned u; } v; v.f = f;
  unsigned r = v.u + 0x7FFFu + ((v.u >> 16) & 1u);
  unsigned short hbits = (unsigned short)(r >> 16);
  __bf16 b;
  __builtin_memcpy(&b, &hbits, sizeof(b));
  return b;
}

static __device__ __forceinline__ v8bf cvt8(v4f a, v4f b) {
  v8bf r;
#pragma unroll
  for (int i = 0; i < 4; ++i) {
    r[i]     = f2bf(a[i]);
    r[i + 4] = f2bf(b[i]);
  }
  return r;
}

static __device__ __forceinline__ float sigmoidf(float x) {
  return 1.0f / (1.0f + __expf(-x));
}

// ---------------------------------------------------------------------------
// Phase 1a: W (4 x [H, C] fp32) -> contiguous bf16 [4][H][C]
// ---------------------------------------------------------------------------
__global__ __launch_bounds__(256)
void cvt_w_kernel(const float* __restrict__ Wi, const float* __restrict__ Wf,
                  const float* __restrict__ Wo, const float* __restrict__ Wg,
                  __bf16* __restrict__ dst)
{
  const size_t base = ((size_t)blockIdx.x * blockDim.x + threadIdx.x) * 8;
  const size_t per = (size_t)HDIM * CDIM;        // elems per gate (mult of 8)
  const float* srcs[4] = { Wi, Wf, Wo, Wg };
  const float* s = srcs[base / per] + (base % per);
  v4f f0 = *(const v4f*)(s);
  v4f f1 = *(const v4f*)(s + 4);
  *(v8bf*)(dst + base) = cvt8(f0, f1);
}

// ---------------------------------------------------------------------------
// Phase 1b: combined = [x | h] fp32 -> bf16 [B][C]
// ---------------------------------------------------------------------------
__global__ __launch_bounds__(256)
void cvt_xh_kernel(const float* __restrict__ x, const float* __restrict__ h,
                   __bf16* __restrict__ dst)
{
  const size_t base = ((size_t)blockIdx.x * blockDim.x + threadIdx.x) * 8;
  const size_t row = base >> 11;          // / CDIM
  const size_t col = base & (CDIM - 1);
  const float* s = (col < IDIM) ? (x + row * IDIM + col)
                                : (h + row * HDIM + (col - IDIM));
  v4f f0 = *(const v4f*)(s);
  v4f f1 = *(const v4f*)(s + 4);
  *(v8bf*)(dst + base) = cvt8(f0, f1);
}

// ---------------------------------------------------------------------------
// Phase 2: pure-bf16 WMMA GEMM + fused LSTM epilogue, gate-split across waves.
// Waves 0-3: gates {i,f}; waves 4-7: gates {o,g}; same 128x32 output tile.
// Each wave: 32 rows x 32 cols x 2 gates = 8 WMMAs / K-step, 4 B fragments
// (1:1 ds:wmma, 2 WMMAs per fragment), 64 accumulator VGPRs.
// o/g accumulators are exchanged through LDS for the elementwise epilogue.
// ---------------------------------------------------------------------------
__global__ __launch_bounds__(256)
void lstm_gemm_bf16(const __bf16* __restrict__ Cbf, const __bf16* __restrict__ Wbf,
                    const float* __restrict__ c,
                    const float* __restrict__ bi, const float* __restrict__ bfv,
                    const float* __restrict__ bo, const float* __restrict__ bg,
                    float* __restrict__ out)
{
  constexpr int LDSROW = 40;                    // padded bf16 row stride (80B)
  constexpr int STAGE_ELEMS = 4 * NT * LDSROW;  // 5120 bf16 per buffer
  // 40960B shared: staging double buffer (20480B) overlaid with epilogue
  // exchange region (32 slots x 320 dwords = 40960B). Staging dead by then.
  __shared__ __align__(16) char lds_raw[40960];
  __bf16* ldsB0 = (__bf16*)lds_raw;
  __bf16* ldsB1 = (__bf16*)lds_raw + STAGE_ELEMS;
  float*  ldsX  = (float*)lds_raw;

  const int tid  = threadIdx.x;
  const int lane = tid & 31;
  const int wave = tid >> 5;
  const int gh   = wave >> 2;      // 0: gates i,f   1: gates o,g
  const int rg   = wave & 3;       // row-group (32 rows each)
  const int m0 = blockIdx.x * MT;
  const int n0 = blockIdx.y * NT;

  // Cooperative W staging: 4*32*32 bf16 per tile, 16 contiguous bf16 per thread.
  const int sg  = tid >> 6;           // gate (wave-uniform)
  const int sn  = (tid & 63) >> 1;    // row in tile
  const int skb = (tid & 1) << 4;     // k elem offset 0 / 16
  const __bf16* wsrc = Wbf + ((size_t)sg * HDIM + (n0 + sn)) * CDIM + skb;
  const int ldoff = (sg * NT + sn) * LDSROW + skb;

  // A fragments: lane half 0 -> K {0..15}; half 1 -> {16..31}
  const int arow = m0 + rg * 32 + (lane & 15);
  const int akc  = (lane >> 4) << 4;
  const __bf16* asrc0 = Cbf + (size_t)arow * CDIM + akc;   // rows +0..15
  const __bf16* asrc1 = asrc0 + (size_t)16 * CDIM;         // rows +16..31

  // B fragment: lane n = l&15, lane half selects 16 contiguous K
  const int boff0 = (lane & 15) * LDSROW + ((lane >> 4) << 4);
  const int gbase = gh * 2;          // first global gate handled by this wave

  v8f acc[2][2][2] = {};   // [gate-local][n-subtile][row-frag]

  // Prologue: stage K-tile 0 into LDS buffer 0.
  {
    v8bf w0 = *(const v8bf*)(wsrc);
    v8bf w1 = *(const v8bf*)(wsrc + 8);
    *(v8bf*)(ldsB0 + ldoff)     = w0;
    *(v8bf*)(ldsB0 + ldoff + 8) = w1;
  }

  constexpr int NKT = CDIM / KT;   // 64
  for (int kt = 0; kt < NKT; ++kt) {
    const int k = kt * KT;
    __syncthreads();
    // Stage next W tile into the other LDS buffer (overlaps with this tile's WMMAs).
    if (kt + 1 < NKT) {
      const __bf16* p = wsrc + (k + KT);
      v8bf w0 = *(const v8bf*)(p);
      v8bf w1 = *(const v8bf*)(p + 8);
      __bf16* d = ((kt & 1) ? ldsB0 : ldsB1) + ldoff;
      *(v8bf*)(d)     = w0;
      *(v8bf*)(d + 8) = w1;
    }

    // Two A fragments: four 16B global loads of bf16.
    v8bf a00 = *(const v8bf*)(asrc0 + k);
    v8bf a01 = *(const v8bf*)(asrc0 + k + 8);
    v8bf a10 = *(const v8bf*)(asrc1 + k);
    v8bf a11 = *(const v8bf*)(asrc1 + k + 8);
    v16bf afrag0, afrag1;
#pragma unroll
    for (int i = 0; i < 8; ++i) {
      afrag0[i] = a00[i]; afrag0[i + 8] = a01[i];
      afrag1[i] = a10[i]; afrag1[i + 8] = a11[i];
    }

    const __bf16* lb = (kt & 1) ? ldsB1 : ldsB0;
#pragma unroll
    for (int gl = 0; gl < 2; ++gl) {
#pragma unroll
      for (int s = 0; s < 2; ++s) {
        const __bf16* bp = &lb[((gbase + gl) * NT + 16 * s) * LDSROW + boff0];
        v8bf lo = *(const v8bf*)(bp);
        v8bf hi = *(const v8bf*)(bp + 8);
        v16bf bfrag;
#pragma unroll
        for (int i = 0; i < 8; ++i) { bfrag[i] = lo[i]; bfrag[i + 8] = hi[i]; }
        acc[gl][s][0] = __builtin_amdgcn_wmma_f32_16x16x32_bf16(
            false, afrag0, false, bfrag, (short)0, acc[gl][s][0], false, false);
        acc[gl][s][1] = __builtin_amdgcn_wmma_f32_16x16x32_bf16(
            false, afrag1, false, bfrag, (short)0, acc[gl][s][1], false, false);
      }
    }
  }

  // ---- Epilogue: exchange o/g accumulators via LDS, then fuse gating. ----
  __syncthreads();   // staging region is dead from here on
  if (gh == 1) {
    // o/g waves publish their tiles: slot = ((rg*2+gl)*2+s)*2+r, 320 dwords/slot,
    // lane-stride 10 dwords (40B) -> conflict-free b64 stores.
#pragma unroll
    for (int gl = 0; gl < 2; ++gl) {
#pragma unroll
      for (int s = 0; s < 2; ++s) {
#pragma unroll
        for (int r = 0; r < 2; ++r) {
          const int slot = ((rg * 2 + gl) * 2 + s) * 2 + r;
          v2f* d = (v2f*)(ldsX + slot * 320 + lane * 10);
          const v8f a = acc[gl][s][r];
          d[0] = (v2f){a[0], a[1]};
          d[1] = (v2f){a[2], a[3]};
          d[2] = (v2f){a[4], a[5]};
          d[3] = (v2f){a[6], a[7]};
        }
      }
    }
  }
  __syncthreads();

  if (gh == 0) {
    const int cbase = n0 + (lane & 15);
    const size_t outC = (size_t)BDIM * HDIM;
#pragma unroll
    for (int s = 0; s < 2; ++s) {
#pragma unroll
      for (int r = 0; r < 2; ++r) {
        // partner tiles: gl=0 -> o_pre, gl=1 -> g_pre (same rg, s, r)
        const int slotO = ((rg * 2 + 0) * 2 + s) * 2 + r;
        const int slotG = ((rg * 2 + 1) * 2 + s) * 2 + r;
        const v2f* po = (const v2f*)(ldsX + slotO * 320 + lane * 10);
        const v2f* pg = (const v2f*)(ldsX + slotG * 320 + lane * 10);
        v8f opre, gpre;
#pragma unroll
        for (int q = 0; q < 4; ++q) {
          v2f o2 = po[q]; opre[2 * q] = o2[0]; opre[2 * q + 1] = o2[1];
          v2f g2 = pg[q]; gpre[2 * q] = g2[0]; gpre[2 * q + 1] = g2[1];
        }
        const int hcol = cbase + 16 * s;
        const float vbi = bi[hcol], vbf = bfv[hcol], vbo = bo[hcol], vbg = bg[hcol];
        const int rbase = m0 + rg * 32 + 16 * r + ((lane >> 4) << 3);
#pragma unroll
        for (int v = 0; v < 8; ++v) {
          const int m = rbase + v;
          const float iv = sigmoidf(acc[0][s][r][v] + vbi);
          const float fv = sigmoidf(acc[1][s][r][v] + vbf);
          const float ov = sigmoidf(opre[v] + vbo);
          const float gv = tanhf(gpre[v] + vbg);
          const float cold = c[(size_t)m * HDIM + hcol];
          const float ct = fv * cold + iv * gv;
          const float ht = ov * tanhf(ct);
          out[(size_t)m * HDIM + hcol] = ht;
          out[outC + (size_t)m * HDIM + hcol] = ct;
        }
      }
    }
  }
}

// ---------------------------------------------------------------------------
// Fallback: fused fp32->bf16-in-loop kernel (used only if workspace too small).
// ---------------------------------------------------------------------------
__global__ __launch_bounds__(256)
void lstm_wmma_fused(const float* __restrict__ x, const float* __restrict__ h,
                     const float* __restrict__ c,
                     const float* __restrict__ Wi, const float* __restrict__ Wf,
                     const float* __restrict__ Wo, const float* __restrict__ Wg,
                     const float* __restrict__ bi, const float* __restrict__ bfv,
                     const float* __restrict__ bo, const float* __restrict__ bg,
                     float* __restrict__ out)
{
  constexpr int LDSROW = 40;
  __shared__ __align__(16) __bf16 ldsB[4 * NT * LDSROW];

  const int tid  = threadIdx.x;
  const int lane = tid & 31;
  const int wave = tid >> 5;
  const int m0 = blockIdx.x * MTF;
  const int n0 = blockIdx.y * NT;

  const float* W4[4] = { Wi, Wf, Wo, Wg };
  const int sg  = tid >> 6;
  const int sn  = (tid & 63) >> 1;
  const int skb = (tid & 1) << 4;
  const float* wsrc_row = W4[sg] + (size_t)(n0 + sn) * CDIM + skb;
  __bf16* ldst = &ldsB[(sg * NT + sn) * LDSROW + skb];

  const int arow = m0 + wave * 16 + (lane & 15);
  const int akc  = (lane >> 4) << 4;
  const int bn   = lane & 15;
  const int bkb  = (lane >> 4) << 4;

  v8f acc[4][2] = {};

  for (int k = 0; k < CDIM; k += KT) {
    __syncthreads();
    {
      const float* p = wsrc_row + k;
      v4f f0 = *(const v4f*)(p + 0);
      v4f f1 = *(const v4f*)(p + 4);
      v4f f2 = *(const v4f*)(p + 8);
      v4f f3 = *(const v4f*)(p + 12);
      *(v8bf*)(ldst)     = cvt8(f0, f1);
      *(v8bf*)(ldst + 8) = cvt8(f2, f3);
    }
    __syncthreads();

    const float* ap = (k < IDIM)
        ? (x + (size_t)arow * IDIM + (k + akc))
        : (h + (size_t)arow * HDIM + (k - IDIM + akc));
    v4f a0 = *(const v4f*)(ap + 0);
    v4f a1 = *(const v4f*)(ap + 4);
    v4f a2 = *(const v4f*)(ap + 8);
    v4f a3 = *(const v4f*)(ap + 12);
    v8bf c0 = cvt8(a0, a1);
    v8bf c1 = cvt8(a2, a3);
    v16bf afrag;
#pragma unroll
    for (int i = 0; i < 8; ++i) { afrag[i] = c0[i]; afrag[i + 8] = c1[i]; }

#pragma unroll
    for (int g = 0; g < 4; ++g) {
#pragma unroll
      for (int s = 0; s < 2; ++s) {
        const __bf16* bp = &ldsB[(g * NT + (16 * s + bn)) * LDSROW + bkb];
        v8bf lo = *(const v8bf*)(bp);
        v8bf hi = *(const v8bf*)(bp + 8);
        v16bf bfrag;
#pragma unroll
        for (int i = 0; i < 8; ++i) { bfrag[i] = lo[i]; bfrag[i + 8] = hi[i]; }
        acc[g][s] = __builtin_amdgcn_wmma_f32_16x16x32_bf16(
            false, afrag, false, bfrag, (short)0, acc[g][s], false, false);
      }
    }
  }

  const int rbase = m0 + wave * 16 + ((lane >> 4) << 3);
  const int cbase = n0 + (lane & 15);
  const size_t outC = (size_t)BDIM * HDIM;
#pragma unroll
  for (int s = 0; s < 2; ++s) {
    const int hcol = cbase + 16 * s;
    const float vbi = bi[hcol], vbf = bfv[hcol], vbo = bo[hcol], vbg = bg[hcol];
#pragma unroll
    for (int v = 0; v < 8; ++v) {
      const int m = rbase + v;
      const float iv = sigmoidf(acc[0][s][v] + vbi);
      const float fv = sigmoidf(acc[1][s][v] + vbf);
      const float ov = sigmoidf(acc[2][s][v] + vbo);
      const float gv = tanhf(acc[3][s][v] + vbg);
      const float cold = c[(size_t)m * HDIM + hcol];
      const float ct = fv * cold + iv * gv;
      const float ht = ov * tanhf(ct);
      out[(size_t)m * HDIM + hcol] = ht;
      out[outC + (size_t)m * HDIM + hcol] = ct;
    }
  }
}

extern "C" void kernel_launch(void* const* d_in, const int* in_sizes, int n_in,
                              void* d_out, int out_size, void* d_ws, size_t ws_size,
                              hipStream_t stream) {
  const float* x  = (const float*)d_in[0];
  const float* h  = (const float*)d_in[1];
  const float* c  = (const float*)d_in[2];
  const float* Wi = (const float*)d_in[3];
  const float* Wf = (const float*)d_in[4];
  const float* Wo = (const float*)d_in[5];
  const float* Wg = (const float*)d_in[6];
  const float* bi = (const float*)d_in[7];
  const float* bf = (const float*)d_in[8];
  const float* bo = (const float*)d_in[9];
  const float* bg = (const float*)d_in[10];
  float* out = (float*)d_out;

  const size_t wsW = (size_t)4 * HDIM * CDIM * sizeof(__bf16);   // 16 MiB
  const size_t wsC = (size_t)BDIM * CDIM * sizeof(__bf16);       // 32 MiB

  if (ws_size >= wsW + wsC) {
    __bf16* Wbf = (__bf16*)d_ws;
    __bf16* Cbf = (__bf16*)((char*)d_ws + wsW);
    const int wThreads = (4 * HDIM * CDIM) / 8;     // 1,048,576
    const int cThreads = (BDIM * CDIM) / 8;         // 2,097,152
    cvt_w_kernel<<<wThreads / 256, 256, 0, stream>>>(Wi, Wf, Wo, Wg, Wbf);
    cvt_xh_kernel<<<cThreads / 256, 256, 0, stream>>>(x, h, Cbf);
    dim3 grid(BDIM / MT, HDIM / NT);  // 64 x 32
    lstm_gemm_bf16<<<grid, 256, 0, stream>>>(Cbf, Wbf, c, bi, bf, bo, bg, out);
  } else {
    dim3 gridF(BDIM / MTF, HDIM / NT);  // 64 x 32
    lstm_wmma_fused<<<gridF, 256, 0, stream>>>(x, h, c, Wi, Wf, Wo, Wg,
                                               bi, bf, bo, bg, out);
  }
  (void)in_sizes; (void)n_in; (void)out_size;
}
